// GATv2Model_80985903333883
// MI455X (gfx1250) — compile-verified, matrix operand order checked by reading, not run on hardware
//
#include <hip/hip_runtime.h>

// ---------------------------------------------------------------------------
// Types for CDNA5 WMMA
// ---------------------------------------------------------------------------
typedef _Float16 v8h  __attribute__((ext_vector_type(8)));
typedef _Float16 v16h __attribute__((ext_vector_type(16)));
typedef float    v8f  __attribute__((ext_vector_type(8)));

// ---------------------------------------------------------------------------
// Helpers
// ---------------------------------------------------------------------------
__device__ __forceinline__ float warp_sum32(float v) {
  #pragma unroll
  for (int o = 16; o > 0; o >>= 1) v += __shfl_xor(v, o, 32);
  return v;
}

// Order-preserving float<->uint key for atomic max
__device__ __forceinline__ unsigned fkey(float f) {
  unsigned u = __float_as_uint(f);
  return (u & 0x80000000u) ? ~u : (u | 0x80000000u);
}
__device__ __forceinline__ float funkey(unsigned k) {
  unsigned u = (k & 0x80000000u) ? (k & 0x7fffffffu) : ~k;
  return __uint_as_float(u);
}

__device__ __forceinline__ void atomic_add_f32(float* p, float v) {
  __hip_atomic_fetch_add(p, v, __ATOMIC_RELAXED, __HIP_MEMORY_SCOPE_AGENT);
}

// ---------------------------------------------------------------------------
// Weight convert + transpose: W (K x M) f32 -> Wt (M x K) f16
// ---------------------------------------------------------------------------
__global__ void k_convT(const float* __restrict__ W, _Float16* __restrict__ Wt,
                        int K, int M) {
  int idx = blockIdx.x * blockDim.x + threadIdx.x;
  if (idx >= K * M) return;
  int k = idx / M, m = idx % M;
  Wt[(size_t)m * K + k] = (_Float16)W[idx];
}

// ---------------------------------------------------------------------------
// Layer 0 projection: x (N x 1): xl[i][m] = x[i]*Wl0[m]+bl0[m] (same for r)
// ---------------------------------------------------------------------------
__global__ void k_l0_proj(const float* __restrict__ x,
                          const float* __restrict__ Wl, const float* __restrict__ bl,
                          const float* __restrict__ Wr, const float* __restrict__ br,
                          float* __restrict__ xl, float* __restrict__ xr, int N) {
  int idx = blockIdx.x * blockDim.x + threadIdx.x;
  if (idx >= N * 256) return;
  int node = idx >> 8, m = idx & 255;
  float xv = x[node];
  xl[idx] = xv * Wl[m] + bl[m];
  xr[idx] = xv * Wr[m] + br[m];
}

// ---------------------------------------------------------------------------
// WMMA GEMM: C (Nrows x M) = A (Nrows x 256, f16 rm) * Bt^T + bias
//   Bt is (M x 256) f16 row-major (i.e. B transposed).
//   One wave -> 16-row tile x 64-col strip (4 accumulators).
// ---------------------------------------------------------------------------
__global__ __launch_bounds__(256)
void k_gemm_wmma(const _Float16* __restrict__ A, const _Float16* __restrict__ Bt,
                 const float* __restrict__ bias, float* __restrict__ C,
                 int nRowTiles, int M) {
  const int K = 256;
  int wave = threadIdx.x >> 5;
  int lane = threadIdx.x & 31;
  int rowTile = blockIdx.y * 8 + wave;
  if (rowTile >= nRowTiles) return;                 // wave-uniform exit
  int colBase = blockIdx.x * 64;
  int rowA = rowTile * 16 + (lane & 15);
  int kb   = (lane >> 4) * 8;                       // lanes 16-31 start at K=8
  int colL = colBase + (lane & 15);

  v8f acc[4] = {v8f{}, v8f{}, v8f{}, v8f{}};
  const _Float16* aRow = A + (size_t)rowA * K;

  #pragma unroll
  for (int k0 = 0; k0 < K; k0 += 32) {
    // A fragment: 16-bit A layout (ISA 7.12.2): lane holds K kb..kb+7, kb+16..kb+23
    v8h alo = *(const v8h*)(aRow + k0 + kb);
    v8h ahi = *(const v8h*)(aRow + k0 + kb + 16);
    v16h a = __builtin_shufflevector(alo, ahi, 0,1,2,3,4,5,6,7,8,9,10,11,12,13,14,15);
    #pragma unroll
    for (int t = 0; t < 4; ++t) {
      const _Float16* bRow = Bt + (size_t)(colL + t * 16) * K + k0 + kb;
      v8h blo = *(const v8h*)(bRow);
      v8h bhi = *(const v8h*)(bRow + 16);
      v16h b = __builtin_shufflevector(blo, bhi, 0,1,2,3,4,5,6,7,8,9,10,11,12,13,14,15);
      acc[t] = __builtin_amdgcn_wmma_f32_16x16x32_f16(
          false, a, false, b, (short)0, acc[t], false, false);
    }
  }

  // D layout: lane 0-15: VGPR r -> (M=r, N=lane); lane 16-31: (M=r+8, N=lane-16)
  int rbase = rowTile * 16 + ((lane >> 4) << 3);
  #pragma unroll
  for (int t = 0; t < 4; ++t) {
    int ncol = colL + t * 16;
    float bv = bias[ncol];
    #pragma unroll
    for (int r = 0; r < 8; ++r)
      C[(size_t)(rbase + r) * M + ncol] = acc[t][r] + bv;
  }
}

// ---------------------------------------------------------------------------
// Edge pass A: logits + segment max.  One wave per edge; C=64 channels/head.
// ---------------------------------------------------------------------------
__global__ void k_edge_logits(const float* __restrict__ xl, const float* __restrict__ xr,
                              const float* __restrict__ att, const int* __restrict__ eidx,
                              int E, int N, int H,
                              float* __restrict__ logit, unsigned* __restrict__ mkey) {
  int gid  = blockIdx.x * blockDim.x + threadIdx.x;
  int edge = gid >> 5;
  int lane = gid & 31;
  int Etot = E + N;
  if (edge >= Etot) return;                          // wave-uniform
  int s = (edge < E) ? eidx[edge]     : edge - E;
  int d = (edge < E) ? eidx[E + edge] : edge - E;
  int F = H * 64;
  for (int h = 0; h < H; ++h) {
    const float* pl = xl + (size_t)s * F + h * 64;
    const float* pr = xr + (size_t)d * F + h * 64;
    const float* pa = att + h * 64;
    float v0 = pl[lane]      + pr[lane];
    float v1 = pl[lane + 32] + pr[lane + 32];
    v0 = v0 > 0.f ? v0 : 0.2f * v0;                  // leaky_relu(0.2)
    v1 = v1 > 0.f ? v1 : 0.2f * v1;
    float part = warp_sum32(v0 * pa[lane] + v1 * pa[lane + 32]);
    if (lane == 0) {
      logit[(size_t)edge * H + h] = part;
      atomicMax(&mkey[(size_t)d * H + h], fkey(part));
    }
  }
}

// ---------------------------------------------------------------------------
// Edge pass B: p = exp(l - m[dst]); atomic-add denominator. Thread/(edge,head).
// ---------------------------------------------------------------------------
__global__ void k_edge_expsum(float* __restrict__ logit, const unsigned* __restrict__ mkey,
                              float* __restrict__ ssum, const int* __restrict__ eidx,
                              int E, int N, int H) {
  int idx = blockIdx.x * blockDim.x + threadIdx.x;
  int Etot = E + N;
  if (idx >= Etot * H) return;
  int edge = idx / H, h = idx - edge * H;
  int d = (edge < E) ? eidx[E + edge] : edge - E;
  float m = funkey(mkey[(size_t)d * H + h]);
  float p = __expf(logit[idx] - m);
  logit[idx] = p;
  atomic_add_f32(&ssum[(size_t)d * H + h], p);
}

// ---------------------------------------------------------------------------
// Edge pass C: out[dst] += (p / s[dst]) * xl[src].  One wave per edge.
// ---------------------------------------------------------------------------
__global__ void k_edge_aggr(const float* __restrict__ xl, const float* __restrict__ p,
                            const float* __restrict__ ssum, const int* __restrict__ eidx,
                            int E, int N, int H, float* __restrict__ out) {
  int gid  = blockIdx.x * blockDim.x + threadIdx.x;
  int edge = gid >> 5;
  int lane = gid & 31;
  int Etot = E + N;
  if (edge >= Etot) return;
  int s = (edge < E) ? eidx[edge]     : edge - E;
  int d = (edge < E) ? eidx[E + edge] : edge - E;
  int F = H * 64;
  for (int h = 0; h < H; ++h) {
    float w = p[(size_t)edge * H + h] / (ssum[(size_t)d * H + h] + 1e-16f);
    const float* pl = xl + (size_t)s * F + h * 64;
    float* po = out + (size_t)d * F + h * 64;
    atomic_add_f32(&po[lane],      w * pl[lane]);
    atomic_add_f32(&po[lane + 32], w * pl[lane + 32]);
  }
}

// ---------------------------------------------------------------------------
// Post: h = elu(layernorm(acc + bias)*g + be); also emit f16 copy for GEMM.
// One block per node, blockDim == F (256 or 64).
// ---------------------------------------------------------------------------
__global__ void k_post_ln_elu(const float* __restrict__ acc, const float* __restrict__ bias,
                              const float* __restrict__ g, const float* __restrict__ be,
                              float* __restrict__ hout, _Float16* __restrict__ h16, int F) {
  int i = blockIdx.x, c = threadIdx.x;
  float v = acc[(size_t)i * F + c] + bias[c];
  __shared__ float sh1[8], sh2[8];
  int w = c >> 5, lane = c & 31, nw = blockDim.x >> 5;
  float s1 = warp_sum32(v), s2 = warp_sum32(v * v);
  if (lane == 0) { sh1[w] = s1; sh2[w] = s2; }
  __syncthreads();
  if (w == 0) {
    float t1 = (lane < nw) ? sh1[lane] : 0.f;
    float t2 = (lane < nw) ? sh2[lane] : 0.f;
    #pragma unroll
    for (int o = 4; o > 0; o >>= 1) { t1 += __shfl_xor(t1, o, 32); t2 += __shfl_xor(t2, o, 32); }
    if (lane == 0) { sh1[0] = t1; sh2[0] = t2; }
  }
  __syncthreads();
  float mean = sh1[0] / F;
  float var  = sh2[0] / F - mean * mean;
  float y = (v - mean) * rsqrtf(var + 1e-5f) * g[c] + be[c];
  y = y > 0.f ? y : (expf(y) - 1.0f);                // elu
  hout[(size_t)i * F + c] = y;
  if (h16) h16[(size_t)i * F + c] = (_Float16)y;
}

// ---------------------------------------------------------------------------
// Final MLP on node 0: gelu(h0 @ rw1 + rb1) @ rw2 + rb2.  One 32-thread block.
// ---------------------------------------------------------------------------
__global__ void k_head_mlp(const float* __restrict__ h0, const float* __restrict__ rw1,
                           const float* __restrict__ rb1, const float* __restrict__ rw2,
                           const float* __restrict__ rb2, float* __restrict__ out) {
  int t = threadIdx.x;                               // 0..31 -> hidden unit
  float a = rb1[t];
  for (int k = 0; k < 64; ++k) a += h0[k] * rw1[k * 32 + t];
  float z = 0.5f * a * (1.0f + erff(a * 0.70710678118654752f));   // exact gelu
  float contrib = warp_sum32(z * rw2[t]);
  if (t == 0) out[0] = contrib + rb2[0];
}

// ---------------------------------------------------------------------------
// Launcher
// ---------------------------------------------------------------------------
extern "C" void kernel_launch(void* const* d_in, const int* in_sizes, int n_in,
                              void* d_out, int out_size, void* d_ws, size_t ws_size,
                              hipStream_t stream) {
  const float* x      = (const float*)d_in[0];
  const int*   eidx   = (const int*)  d_in[1];
  const float* Wl0    = (const float*)d_in[2];  const float* bl0 = (const float*)d_in[3];
  const float* Wr0    = (const float*)d_in[4];  const float* br0 = (const float*)d_in[5];
  const float* att0   = (const float*)d_in[6];  const float* bias0 = (const float*)d_in[7];
  const float* Wl1    = (const float*)d_in[8];  const float* bl1 = (const float*)d_in[9];
  const float* Wr1    = (const float*)d_in[10]; const float* br1 = (const float*)d_in[11];
  const float* att1   = (const float*)d_in[12]; const float* bias1 = (const float*)d_in[13];
  const float* Wl2    = (const float*)d_in[14]; const float* bl2 = (const float*)d_in[15];
  const float* Wr2    = (const float*)d_in[16]; const float* br2 = (const float*)d_in[17];
  const float* att2   = (const float*)d_in[18]; const float* bias2 = (const float*)d_in[19];
  const float* g0 = (const float*)d_in[20]; const float* be0 = (const float*)d_in[21];
  const float* g1 = (const float*)d_in[22]; const float* be1 = (const float*)d_in[23];
  const float* g2 = (const float*)d_in[24]; const float* be2 = (const float*)d_in[25];
  const float* rw1 = (const float*)d_in[26]; const float* rb1 = (const float*)d_in[27];
  const float* rw2 = (const float*)d_in[28]; const float* rb2 = (const float*)d_in[29];

  const int N = in_sizes[0];          // 50000 (x is N x 1)
  const int E = in_sizes[1] / 2;      // 800000
  const int Etot = E + N;             // with self loops

  // --- workspace carve-out ---
  char* base = (char*)d_ws;
  size_t off = 0;
  auto take = [&](size_t bytes) -> char* {
    char* p = base + off;
    off = (off + bytes + 255) & ~(size_t)255;
    return p;
  };
  float*     h     = (float*)    take((size_t)N * 256 * 4);
  _Float16*  h16   = (_Float16*) take((size_t)N * 256 * 2);
  float*     xl    = (float*)    take((size_t)N * 256 * 4);
  float*     xr    = (float*)    take((size_t)N * 256 * 4);
  float*     accum = (float*)    take((size_t)N * 256 * 4);
  float*     logit = (float*)    take((size_t)Etot * 4 * 4);
  unsigned*  mkey  = (unsigned*) take((size_t)N * 4 * 4);
  float*     ssum  = (float*)    take((size_t)N * 4 * 4);
  _Float16*  W1lt  = (_Float16*) take(256 * 256 * 2);
  _Float16*  W1rt  = (_Float16*) take(256 * 256 * 2);
  _Float16*  W2lt  = (_Float16*) take(64 * 256 * 2);
  _Float16*  W2rt  = (_Float16*) take(64 * 256 * 2);
  (void)ws_size;

  const int nRowTiles = N / 16;       // 3125 exact for N=50000
  const int edgeBlocks = (Etot * 32 + 255) / 256;

  // --- weight conversion (every call; no caching) ---
  k_convT<<<(256 * 256 + 255) / 256, 256, 0, stream>>>(Wl1, W1lt, 256, 256);
  k_convT<<<(256 * 256 + 255) / 256, 256, 0, stream>>>(Wr1, W1rt, 256, 256);
  k_convT<<<(256 * 64 + 255) / 256, 256, 0, stream>>>(Wl2, W2lt, 256, 64);
  k_convT<<<(256 * 64 + 255) / 256, 256, 0, stream>>>(Wr2, W2rt, 256, 64);

  auto edge_phase = [&](const float* XL, const float* XR, const float* ATT,
                        int H, int F) {
    hipMemsetAsync(mkey,  0, (size_t)N * H * 4, stream);        // key 0 == -inf
    hipMemsetAsync(ssum,  0, (size_t)N * H * 4, stream);
    hipMemsetAsync(accum, 0, (size_t)N * F * 4, stream);
    k_edge_logits<<<edgeBlocks, 256, 0, stream>>>(XL, XR, ATT, eidx, E, N, H, logit, mkey);
    k_edge_expsum<<<(Etot * H + 255) / 256, 256, 0, stream>>>(logit, mkey, ssum, eidx, E, N, H);
    k_edge_aggr<<<edgeBlocks, 256, 0, stream>>>(XL, logit, ssum, eidx, E, N, H, accum);
  };

  // ---------------- Layer 0: in=1, 4 heads x 64, concat ----------------
  k_l0_proj<<<((size_t)N * 256 + 255) / 256, 256, 0, stream>>>(x, Wl0, bl0, Wr0, br0, xl, xr, N);
  edge_phase(xl, xr, att0, 4, 256);
  k_post_ln_elu<<<N, 256, 0, stream>>>(accum, bias0, g0, be0, h, h16, 256);

  // ---------------- Layer 1: 256 -> 4 heads x 64, concat ----------------
  {
    dim3 grid(256 / 64, (nRowTiles + 7) / 8);
    k_gemm_wmma<<<grid, 256, 0, stream>>>(h16, W1lt, bl1, xl, nRowTiles, 256);
    k_gemm_wmma<<<grid, 256, 0, stream>>>(h16, W1rt, br1, xr, nRowTiles, 256);
  }
  edge_phase(xl, xr, att1, 4, 256);
  k_post_ln_elu<<<N, 256, 0, stream>>>(accum, bias1, g1, be1, h, h16, 256);

  // ---------------- Layer 2: 256 -> 1 head x 64, mean(=identity) --------
  {
    dim3 grid(64 / 64, (nRowTiles + 7) / 8);
    k_gemm_wmma<<<grid, 256, 0, stream>>>(h16, W2lt, bl2, xl, nRowTiles, 64);
    k_gemm_wmma<<<grid, 256, 0, stream>>>(h16, W2rt, br2, xr, nRowTiles, 64);
  }
  edge_phase(xl, xr, att2, 1, 64);
  k_post_ln_elu<<<N, 64, 0, stream>>>(accum, bias2, g2, be2, h, (_Float16*)nullptr, 64);

  // ---------------- Readout MLP on node 0 ----------------
  k_head_mlp<<<1, 32, 0, stream>>>(h, rw1, rb1, rw2, rb2, (float*)d_out);
}
